// Masked_Attention_18416819765633
// MI455X (gfx1250) — matrix operand order repaired and checked by reading, not compiled, not run on hardware
//
#include <hip/hip_runtime.h>
#include <hip/hip_bf16.h>
#include <cstddef>

// ---------------------------------------------------------------------------
// Masked axial attention (AST-style) for MI455X / gfx1250.
// All matmuls run on v_wmma_f32_16x16x32_bf16 with f32 accumulation.
// v3: software-pipelined (double-buffered) fragment loads so WMMAs wait on
//     the *previous* K-step's loads (partial s_wait_loadcnt), not the current.
// ---------------------------------------------------------------------------

#define BATCH   16
#define NTOK    1213          // 12*101 + 1 (CLS)
#define NPAD    1216          // NTOK padded to 16
#define MTILES  (NPAD / 16)   // 76
#define DIM     768
#define NHEADS  12
#define HD      64
#define THREEC  (3 * DIM)     // 2304
#define TDIMC   101
#define SPITCH  (NPAD + 8)    // LDS row pitch (floats); 4896B rows, 32B aligned

typedef __attribute__((ext_vector_type(16))) __bf16 v16bf;
typedef __attribute__((ext_vector_type(8)))  __bf16 v8bf;
typedef __attribute__((ext_vector_type(8)))  float  v8f;

// D = A(16x32 bf16) x B(32x16 bf16) + C(16x16 f32)
static __device__ inline v8f wmma_bf16(v16bf a, v16bf b, v8f c) {
  return __builtin_amdgcn_wmma_f32_16x16x32_bf16(
      /*neg_a=*/false, a, /*neg_b=*/false, b,
      /*c_mod=*/(short)0, c, /*reuse_a=*/false, /*reuse_b=*/false);
}

// A-fragment (16x32, bf16) from row-major [rows][ld].
// ISA layout: lane<16 -> row=lane,    K = {kb..kb+7, kb+16..kb+23}
//             lane>=16 -> row=lane-16, K = {kb+8..kb+15, kb+24..kb+31}
static __device__ inline v16bf load_frag_A(const __bf16* __restrict__ base,
                                           int ld, int row0, int kb, int lane) {
  const int r    = row0 + (lane & 15);
  const int half = lane >> 4;
  const v8bf* p0 = (const v8bf*)(base + (size_t)r * ld + kb + half * 8);
  const v8bf* p1 = (const v8bf*)(base + (size_t)r * ld + kb + 16 + half * 8);
  v8bf lo = *p0, hi = *p1;
  return __builtin_shufflevector(lo, hi, 0, 1, 2, 3, 4, 5, 6, 7,
                                 8, 9, 10, 11, 12, 13, 14, 15);
}

// B-fragment (32x16, bf16): lane n holds column n of B, i.e. row (row0+n) of
// the [Nout][K] row-major operand. lane<16 -> K = {kb..kb+7, kb+16..kb+23},
// lane>=16 -> K = {kb+8..kb+15, kb+24..kb+31} (same order as A).
static __device__ inline v16bf load_frag_B(const __bf16* __restrict__ base,
                                           int ld, int row0, int kb, int lane) {
  const int n    = row0 + (lane & 15);
  const int half = lane >> 4;
  const v8bf* p0 = (const v8bf*)(base + (size_t)n * ld + kb + half * 8);
  v8bf lo = p0[0], hi = p0[2];
  return __builtin_shufflevector(lo, hi, 0, 1, 2, 3, 4, 5, 6, 7,
                                 8, 9, 10, 11, 12, 13, 14, 15);
}

// ---------------------------------------------------------------------------
// Cast kernels: f32 -> bf16 workspace copies
// ---------------------------------------------------------------------------
__global__ void cast_x_kernel(const float* __restrict__ x,
                              __bf16* __restrict__ xb) {
  size_t idx = (size_t)blockIdx.x * blockDim.x + threadIdx.x;
  const size_t total = (size_t)BATCH * NPAD * DIM;
  if (idx >= total) return;
  int    c = (int)(idx % DIM);
  size_t t = idx / DIM;
  int    n = (int)(t % NPAD);
  int    b = (int)(t / NPAD);
  float v = (n < NTOK) ? x[((size_t)b * NTOK + n) * DIM + c] : 0.0f;
  xb[idx] = (__bf16)v;
}

__global__ void cast_w_kernel(const float* __restrict__ w,
                              __bf16* __restrict__ wb, size_t count) {
  size_t idx = (size_t)blockIdx.x * blockDim.x + threadIdx.x;
  if (idx >= count) return;
  wb[idx] = (__bf16)w[idx];
}

// ---------------------------------------------------------------------------
// QKV GEMM: [NPAD,768] x [2304,768]^T + bias, split into q/k (row-major per
// head) and v stored TRANSPOSED per head (so PV B-fragments are row loads).
// Each wave computes a 16x64 output panel. Fragment loads are double-buffered:
// step k's WMMAs consume registers loaded at step k-1.
// ---------------------------------------------------------------------------
__global__ void qkv_gemm_kernel(const __bf16* __restrict__ xb,   // [B][NPAD][DIM]
                                const __bf16* __restrict__ wb,   // [THREEC][DIM]
                                const float*  __restrict__ bias, // [THREEC]
                                __bf16* __restrict__ qbuf,       // [B][H][NPAD][HD]
                                __bf16* __restrict__ kbuf,       // [B][H][NPAD][HD]
                                __bf16* __restrict__ vtb) {      // [B][H][HD][NPAD]
  const int lane  = threadIdx.x;
  const int og    = blockIdx.x * blockDim.y + threadIdx.y; // 0..35, group of 4 otiles
  const int mtile = blockIdx.y;                            // 0..75
  const int b     = blockIdx.z;
  const __bf16* A  = xb + (size_t)b * NPAD * DIM;
  const int otile0 = og * 4;
  const int m0     = mtile * 16;

  v8f acc0 = {}, acc1 = {}, acc2 = {}, acc3 = {};

  // Prologue: load K-step 0 fragments.
  v16bf a_c  = load_frag_A(A,  DIM, m0, 0, lane);
  v16bf w0_c = load_frag_B(wb, DIM, (otile0 + 0) * 16, 0, lane);
  v16bf w1_c = load_frag_B(wb, DIM, (otile0 + 1) * 16, 0, lane);
  v16bf w2_c = load_frag_B(wb, DIM, (otile0 + 2) * 16, 0, lane);
  v16bf w3_c = load_frag_B(wb, DIM, (otile0 + 3) * 16, 0, lane);

  for (int kc = 0; kc < DIM; kc += 32) {
    const int kn = kc + 32;
    v16bf a_n = a_c, w0_n = w0_c, w1_n = w1_c, w2_n = w2_c, w3_n = w3_c;
    if (kn < DIM) {
      __builtin_prefetch(A + (size_t)(m0 + (lane & 15)) * DIM + kn, 0, 1);
      a_n  = load_frag_A(A,  DIM, m0, kn, lane);
      w0_n = load_frag_B(wb, DIM, (otile0 + 0) * 16, kn, lane);
      w1_n = load_frag_B(wb, DIM, (otile0 + 1) * 16, kn, lane);
      w2_n = load_frag_B(wb, DIM, (otile0 + 2) * 16, kn, lane);
      w3_n = load_frag_B(wb, DIM, (otile0 + 3) * 16, kn, lane);
    }
    acc0 = wmma_bf16(a_c, w0_c, acc0);
    acc1 = wmma_bf16(a_c, w1_c, acc1);
    acc2 = wmma_bf16(a_c, w2_c, acc2);
    acc3 = wmma_bf16(a_c, w3_c, acc3);
    a_c = a_n; w0_c = w0_n; w1_c = w1_n; w2_c = w2_n; w3_c = w3_n;
  }

  const int col  = lane & 15;
  const int half = lane >> 4;
#pragma unroll
  for (int t = 0; t < 4; ++t) {
    const v8f acc = (t == 0) ? acc0 : (t == 1) ? acc1 : (t == 2) ? acc2 : acc3;
    const int o     = (otile0 + t) * 16 + col;
    const float bv  = bias[o];
    const int which = o / DIM;
    const int oc    = o % DIM;
    const int h     = oc / HD;
    const int d     = oc % HD;
#pragma unroll
    for (int r = 0; r < 8; ++r) {
      const int m = m0 + r + half * 8;
      const float v = acc[r] + bv;
      if (which == 0)
        qbuf[(((size_t)b * NHEADS + h) * NPAD + m) * HD + d] = (__bf16)v;
      else if (which == 1)
        kbuf[(((size_t)b * NHEADS + h) * NPAD + m) * HD + d] = (__bf16)v;
      else
        vtb[(((size_t)b * NHEADS + h) * HD + d) * NPAD + m] = (__bf16)v;
    }
  }
}

// ---------------------------------------------------------------------------
// Attention: one block per (b, h, query-tile). Full 16 x NPAD f32 score panel
// lives in LDS (~78 KB; CDNA5 WGP has 320 KB). 4 waves.
//   Phase 1: scores via WMMA (Q hoisted, K fragments double-buffered)
//   Phase 2: row softmax stats (shfl_xor trees over 8-lane groups)
//   Phase 3: O = P @ V, P from LDS via v8f (ds_load_b128) -> bf16,
//            V^T fragments double-buffered.
// ---------------------------------------------------------------------------
__global__ void attn_kernel(const __bf16* __restrict__ q,   // [B][H][NPAD][HD]
                            const __bf16* __restrict__ k,   // [B][H][NPAD][HD]
                            const __bf16* __restrict__ vt,  // [B][H][HD][NPAD]
                            __bf16* __restrict__ obuf) {    // [B][NPAD][DIM]
  extern __shared__ float smem[];
  float* S       = smem;                 // [16][SPITCH]
  float* rowstat = smem + 16 * SPITCH;   // [16] -> 1/rowsum

  const int lane = threadIdx.x;
  const int wave = threadIdx.y;          // 0..3
  const int tid  = wave * 32 + lane;
  const int qt   = blockIdx.x;           // 0..75
  const int h    = blockIdx.y;
  const int b    = blockIdx.z;

  const __bf16* Q  = q  + ((size_t)b * NHEADS + h) * NPAD * HD;
  const __bf16* K  = k  + ((size_t)b * NHEADS + h) * NPAD * HD;
  const __bf16* VT = vt + ((size_t)b * NHEADS + h) * HD * NPAD;

  const float scale = 0.125f;            // 64^-0.5
  const int col  = lane & 15;
  const int half = lane >> 4;

  // ---- Phase 1: masked scaled scores into LDS ----
  {
    // Q tile is invariant across key tiles: load its two fragments once.
    const v16bf qa0 = load_frag_A(Q, HD, qt * 16, 0,  lane);
    const v16bf qa1 = load_frag_A(Q, HD, qt * 16, 32, lane);

    v16bf kb0_c = load_frag_B(K, HD, wave * 16, 0,  lane);
    v16bf kb1_c = load_frag_B(K, HD, wave * 16, 32, lane);
    for (int jt = wave; jt < MTILES; jt += 4) {
      const int jn = jt + 4;
      v16bf kb0_n = kb0_c, kb1_n = kb1_c;
      if (jn < MTILES) {
        kb0_n = load_frag_B(K, HD, jn * 16, 0,  lane);
        kb1_n = load_frag_B(K, HD, jn * 16, 32, lane);
      }
      v8f acc = {};
      acc = wmma_bf16(qa0, kb0_c, acc);
      acc = wmma_bf16(qa1, kb1_c, acc);
      kb0_c = kb0_n; kb1_c = kb1_n;

      const int  kt     = jt * 16 + col;         // key token for this lane
      const bool kvalid = (kt < NTOK);
      const int  kik    = (kt - 1) / TDIMC;
      const int  kjk    = (kt - 1) % TDIMC;
#pragma unroll
      for (int r = 0; r < 8; ++r) {
        const int m    = r + half * 8;           // row within tile
        const int qtok = qt * 16 + m;
        float sv;
        if (!kvalid) {
          sv = -3.0e38f;                         // padded key -> exp() = 0
        } else {
          const int iq = (qtok - 1) / TDIMC;
          const int jq = (qtok - 1) % TDIMC;
          const bool mk = (qtok == 0) || (kt == 0) || (iq == kik) || (jq == kjk);
          sv = mk ? acc[r] * scale : 0.0f;       // multiplicative 0/1 mask
        }
        S[m * SPITCH + jt * 16 + col] = sv;
      }
    }
  }
  __syncthreads();

  // ---- Phase 2: softmax stats, 8 threads per row ----
  {
    const int row = tid >> 3;                    // 0..15
    const int sub = tid & 7;
    float mx = -3.0e38f;
    for (int c = sub; c < NTOK; c += 8) mx = fmaxf(mx, S[row * SPITCH + c]);
#pragma unroll
    for (int d = 1; d < 8; d <<= 1) mx = fmaxf(mx, __shfl_xor(mx, d, 32));
    float sum = 0.0f;
    for (int c = sub; c < NPAD; c += 8) {
      const float e = __expf(S[row * SPITCH + c] - mx);
      S[row * SPITCH + c] = e;                   // padded cols -> 0
      sum += e;
    }
#pragma unroll
    for (int d = 1; d < 8; d <<= 1) sum += __shfl_xor(sum, d, 32);
    if (sub == 0) rowstat[row] = 1.0f / sum;
  }
  __syncthreads();

  // ---- Phase 3: O = P @ V. Each wave owns one 16-wide slice of hd. ----
  {
    const float inv = rowstat[col];              // A-frag row == (lane & 15)
    const int   d0  = wave * 16;
    const float* rp = S + (size_t)col * SPITCH;  // row base, 32B aligned
    v8f acc = {};
    v16bf bb_c = load_frag_B(VT, NPAD, d0, 0, lane);
    for (int kc = 0; kc < NPAD; kc += 32) {
      const int kn = kc + 32;
      v16bf bb_n = bb_c;
      if (kn < NPAD) bb_n = load_frag_B(VT, NPAD, d0, kn, lane);

      const v8f f0 = *(const v8f*)(rp + kc + half * 8);
      const v8f f1 = *(const v8f*)(rp + kc + 16 + half * 8);
      v16bf a;
#pragma unroll
      for (int e = 0; e < 8; ++e) a[e]     = (__bf16)(f0[e] * inv);
#pragma unroll
      for (int e = 0; e < 8; ++e) a[e + 8] = (__bf16)(f1[e] * inv);

      acc = wmma_bf16(a, bb_c, acc);
      bb_c = bb_n;
    }
#pragma unroll
    for (int r = 0; r < 8; ++r) {
      const int m = qt * 16 + r + half * 8;
      obuf[((size_t)b * NPAD + m) * DIM + h * HD + d0 + col] = (__bf16)acc[r];
    }
  }
}

// ---------------------------------------------------------------------------
// Output projection: [NPAD,768] x [768,768]^T + bias -> f32 d_out [B][NTOK][768]
// Each wave computes a 16x64 output panel; fragments double-buffered.
// ---------------------------------------------------------------------------
__global__ void proj_gemm_kernel(const __bf16* __restrict__ ob,  // [B][NPAD][DIM]
                                 const __bf16* __restrict__ pw,  // [DIM][DIM]
                                 const float*  __restrict__ pb,  // [DIM]
                                 float* __restrict__ out) {      // [B][NTOK][DIM]
  const int lane  = threadIdx.x;
  const int og    = blockIdx.x * blockDim.y + threadIdx.y; // 0..11
  const int mtile = blockIdx.y;                            // 0..75
  const int b     = blockIdx.z;
  const __bf16* A  = ob + (size_t)b * NPAD * DIM;
  const int otile0 = og * 4;
  const int m0     = mtile * 16;

  v8f acc0 = {}, acc1 = {}, acc2 = {}, acc3 = {};

  v16bf a_c  = load_frag_A(A,  DIM, m0, 0, lane);
  v16bf w0_c = load_frag_B(pw, DIM, (otile0 + 0) * 16, 0, lane);
  v16bf w1_c = load_frag_B(pw, DIM, (otile0 + 1) * 16, 0, lane);
  v16bf w2_c = load_frag_B(pw, DIM, (otile0 + 2) * 16, 0, lane);
  v16bf w3_c = load_frag_B(pw, DIM, (otile0 + 3) * 16, 0, lane);

  for (int kc = 0; kc < DIM; kc += 32) {
    const int kn = kc + 32;
    v16bf a_n = a_c, w0_n = w0_c, w1_n = w1_c, w2_n = w2_c, w3_n = w3_c;
    if (kn < DIM) {
      __builtin_prefetch(A + (size_t)(m0 + (lane & 15)) * DIM + kn, 0, 1);
      a_n  = load_frag_A(A,  DIM, m0, kn, lane);
      w0_n = load_frag_B(pw, DIM, (otile0 + 0) * 16, kn, lane);
      w1_n = load_frag_B(pw, DIM, (otile0 + 1) * 16, kn, lane);
      w2_n = load_frag_B(pw, DIM, (otile0 + 2) * 16, kn, lane);
      w3_n = load_frag_B(pw, DIM, (otile0 + 3) * 16, kn, lane);
    }
    acc0 = wmma_bf16(a_c, w0_c, acc0);
    acc1 = wmma_bf16(a_c, w1_c, acc1);
    acc2 = wmma_bf16(a_c, w2_c, acc2);
    acc3 = wmma_bf16(a_c, w3_c, acc3);
    a_c = a_n; w0_c = w0_n; w1_c = w1_n; w2_c = w2_n; w3_c = w3_n;
  }

  const int col  = lane & 15;
  const int half = lane >> 4;
#pragma unroll
  for (int t = 0; t < 4; ++t) {
    const v8f acc  = (t == 0) ? acc0 : (t == 1) ? acc1 : (t == 2) ? acc2 : acc3;
    const int o    = (otile0 + t) * 16 + col;
    const float bv = pb[o];
#pragma unroll
    for (int r = 0; r < 8; ++r) {
      const int m = m0 + r + half * 8;
      if (m < NTOK) out[((size_t)b * NTOK + m) * DIM + o] = acc[r] + bv;
    }
  }
}

// ---------------------------------------------------------------------------
extern "C" void kernel_launch(void* const* d_in, const int* in_sizes, int n_in,
                              void* d_out, int out_size, void* d_ws, size_t ws_size,
                              hipStream_t stream) {
  const float* x      = (const float*)d_in[0];
  const float* qkv_w  = (const float*)d_in[1];
  const float* qkv_b  = (const float*)d_in[2];
  const float* proj_w = (const float*)d_in[3];
  const float* proj_b = (const float*)d_in[4];
  float* out = (float*)d_out;

  // Workspace carve-up (bf16 buffers), ~154 MB total.
  char* ws = (char*)d_ws;
  __bf16* xb   = (__bf16*)ws; ws += (size_t)BATCH * NPAD * DIM * sizeof(__bf16);
  __bf16* wqkv = (__bf16*)ws; ws += (size_t)THREEC * DIM * sizeof(__bf16);
  __bf16* pw   = (__bf16*)ws; ws += (size_t)DIM * DIM * sizeof(__bf16);
  __bf16* qb   = (__bf16*)ws; ws += (size_t)BATCH * NHEADS * NPAD * HD * sizeof(__bf16);
  __bf16* kb   = (__bf16*)ws; ws += (size_t)BATCH * NHEADS * NPAD * HD * sizeof(__bf16);
  __bf16* vtb  = (__bf16*)ws; ws += (size_t)BATCH * NHEADS * HD * NPAD * sizeof(__bf16);
  __bf16* obuf = (__bf16*)ws; ws += (size_t)BATCH * NPAD * DIM * sizeof(__bf16);

  // 1) Cast inputs to bf16 (x is zero-padded to NPAD rows).
  {
    const size_t total = (size_t)BATCH * NPAD * DIM;
    cast_x_kernel<<<dim3((unsigned)((total + 255) / 256)), dim3(256), 0, stream>>>(x, xb);
    const size_t wq = (size_t)THREEC * DIM;
    cast_w_kernel<<<dim3((unsigned)((wq + 255) / 256)), dim3(256), 0, stream>>>(qkv_w, wqkv, wq);
    const size_t wp = (size_t)DIM * DIM;
    cast_w_kernel<<<dim3((unsigned)((wp + 255) / 256)), dim3(256), 0, stream>>>(proj_w, pw, wp);
  }

  // 2) QKV projection (WMMA, 16x64 per wave): 36 ogroups x 76 mtiles x 16 b.
  qkv_gemm_kernel<<<dim3(THREEC / 16 / 4 / 4, MTILES, BATCH), dim3(32, 4), 0, stream>>>(
      xb, wqkv, qkv_b, qb, kb, vtb);

  // 3) Masked attention (WMMA + LDS score panel + softmax).
  {
    const size_t lds_bytes = (size_t)(16 * SPITCH + 32) * sizeof(float); // ~78.5 KB
    attn_kernel<<<dim3(MTILES, NHEADS, BATCH), dim3(32, 4), lds_bytes, stream>>>(
        qb, kb, vtb, obuf);
  }

  // 4) Output projection (WMMA, 16x64 per wave) -> f32 result.
  proj_gemm_kernel<<<dim3(DIM / 16 / 4 / 4, MTILES, BATCH), dim3(32, 4), 0, stream>>>(
      obuf, pw, proj_b, out);
}
